// GPT_91259465105715
// MI455X (gfx1250) — matrix operand order repaired and checked by reading, not compiled, not run on hardware
//
#include <hip/hip_runtime.h>
#include <stdint.h>

// ---------- CDNA5 WMMA types ----------
typedef __attribute__((ext_vector_type(16))) __bf16 v16bf;
typedef __attribute__((ext_vector_type(8)))  float  v8f;
typedef __attribute__((ext_vector_type(4)))  int    v4i;

#define T_SEQ  2048
#define DMODEL 1024
#define NHEAD  16
#define HDIM   64
#define NROWS  4096   // B*T

#define AS1 __attribute__((address_space(1)))
#define AS3 __attribute__((address_space(3)))

#if defined(__has_builtin)
#if __has_builtin(__builtin_amdgcn_global_load_async_to_lds_b128)
#define HAVE_ASYNC_LDS_BUILTIN 1
#endif
#if __has_builtin(__builtin_amdgcn_s_wait_asynccnt)
#define HAVE_WAIT_ASYNC_BUILTIN 1
#endif
#endif

union Frag {                 // one lane's 16 bf16 elements (A or B fragment)
  uint4 u[2];
  v16bf v;
  unsigned short s[16];
};
union Acc {                  // one lane's 8 f32 C/D elements
  v8f   v;
  float f[8];
};

__device__ inline unsigned short f2bf(float x) {   // round-to-nearest-even f32->bf16
  unsigned int u = __float_as_uint(x);
  u += 0x7FFFu + ((u >> 16) & 1u);
  return (unsigned short)(u >> 16);
}
__device__ inline v8f zero8() {
  v8f z = {0.f, 0.f, 0.f, 0.f, 0.f, 0.f, 0.f, 0.f};
  return z;
}

// ---- async global->LDS copy (ASYNCcnt path, CDNA5 ISA) ----
template <int OFF>
__device__ inline void async_copy_b128(const unsigned short* g, unsigned short* l) {
#ifdef HAVE_ASYNC_LDS_BUILTIN
  __builtin_amdgcn_global_load_async_to_lds_b128(
      (AS1 v4i*)(g), (AS3 v4i*)(l), OFF, 0);
#else
  asm volatile("global_load_async_to_lds_b128 %0, %1, off offset:%c2"
               :: "v"((unsigned)(uintptr_t)(AS3 void*)(l)), "v"(g), "i"(OFF)
               : "memory");
#endif
}
__device__ inline void wait_async0() {
#ifdef HAVE_WAIT_ASYNC_BUILTIN
  __builtin_amdgcn_s_wait_asynccnt(0);
#else
  asm volatile("s_wait_asynccnt 0x0" ::: "memory");
#endif
}

enum { GF_BIAS = 1, GF_RELU = 2, GF_RES = 4, GF_OUTF = 8, GF_OUTB = 16, GF_VT = 32 };

// ---------- generic bf16 WMMA GEMM:  C[M,N] = A[M,K] @ W[K,N] (+bias,+res,relu) ----------
// A: bf16 row-major [M,K] (direct global loads: rows are wave-private, no cross-wave reuse).
// Bt: bf16 [N,K] transposed weight; the 64x64 B tile is shared by all 8 waves, so it is
// staged through LDS with double-buffered GLOBAL_LOAD_ASYNC_TO_LDS_B128 (ASYNCcnt).
// Block: 256 threads = 8 waves; block tile 128(M) x 64(N); wave w owns rows w*16..w*16+15.
__global__ __launch_bounds__(256) void gemm_bf16(
    const unsigned short* __restrict__ A, const unsigned short* __restrict__ Bt,
    const float* __restrict__ bias, float* __restrict__ outF,
    unsigned short* __restrict__ outB, int M, int N, int K, int flags)
{
  __shared__ unsigned short Blds[2][64 * 64];   // [buf][n][k] 8KB each

  const int tid  = threadIdx.x;
  const int lane = tid & 31;
  const int wave = tid >> 5;
  const int l15  = lane & 15;
  const int hh   = lane >> 4;                 // lane half: 0 or 1
  const int rBase = blockIdx.y * 128 + wave * 16;
  const int cBase = blockIdx.x * 64;

  const unsigned short* Arow = A + (size_t)(rBase + l15) * K;

  // async staging coordinates: thread tid moves 32B of row (cBase + tid/4)
  const int brow = tid >> 2;                  // 0..63  (n within tile)
  const int bcol = (tid & 3) * 16;            // 0,16,32,48 (k within tile, elems)
  const unsigned short* Bsrc = Bt + (size_t)(cBase + brow) * K + bcol;

  Acc c[4];
#pragma unroll
  for (int j = 0; j < 4; ++j) c[j].v = zero8();

  // prologue: stage tile 0 into buffer 0
  {
    unsigned short* d = &Blds[0][brow * 64 + bcol];
    async_copy_b128<0>(Bsrc, d);
    async_copy_b128<16>(Bsrc, d);
  }

  const int NT = K >> 6;                      // K/64 tiles
  for (int kt = 0; kt < NT; ++kt) {
    const int cur = kt & 1;
    wait_async0();                            // own async stores to LDS done
    __syncthreads();                          // all waves' tile-kt data visible
    if (kt + 1 < NT) {                        // prefetch next tile into other buffer
      const unsigned short* s = Bsrc + (size_t)(kt + 1) * 64;
      unsigned short* d = &Blds[cur ^ 1][brow * 64 + bcol];
      async_copy_b128<0>(s, d);
      async_copy_b128<16>(s, d);
    }
    const int k0 = kt * 64;
#pragma unroll
    for (int ks = 0; ks < 2; ++ks) {
      // A fragment (16x32 16-bit layout): lane row = l15, K chunks at hh*8 and 16+hh*8
      Frag a;
      a.u[0] = *(const uint4*)(Arow + k0 + ks * 32 + hh * 8);
      a.u[1] = *(const uint4*)(Arow + k0 + ks * 32 + hh * 8 + 16);
#pragma unroll
      for (int j = 0; j < 4; ++j) {
        // B fragment (32x16): lane col = l15, K chunks at hh*8 and 16+hh*8 — from LDS
        const unsigned short* bp = &Blds[cur][(j * 16 + l15) * 64 + ks * 32 + hh * 8];
        Frag b;
        b.u[0] = *(const uint4*)(bp);
        b.u[1] = *(const uint4*)(bp + 16);
        c[j].v = __builtin_amdgcn_wmma_f32_16x16x32_bf16(
            false, a.v, false, b.v, (short)0, c[j].v, false, false);
      }
    }
    __syncthreads();                          // all reads of Blds[cur] done before reuse
  }

#pragma unroll
  for (int j = 0; j < 4; ++j) {
#pragma unroll
    for (int r = 0; r < 8; ++r) {
      const int m = rBase + r + hh * 8;       // C layout: row = r + 8*half
      const int n = cBase + j * 16 + l15;     // col = lane%16
      float val = c[j].f[r];
      const size_t off = (size_t)m * N + n;
      if (flags & GF_BIAS) val += bias[n];
      if (flags & GF_RES)  val += outF[off];
      if (flags & GF_RELU) val = fmaxf(val, 0.0f);
      if (flags & GF_OUTF) outF[off] = val;
      if (flags & GF_VT) {
        // scatter V as [b,h,e,T] bf16 so attention B-fragments are contiguous
        const int bb = m >> 11, tt = m & (T_SEQ - 1);
        const int hd = n >> 6,  ee = n & (HDIM - 1);
        outB[(size_t)((bb * NHEAD + hd) * HDIM + ee) * T_SEQ + tt] = f2bf(val);
      } else if (flags & GF_OUTB) {
        outB[off] = f2bf(val);
      }
    }
  }
}

// ---------- flash attention: 1 wave per (16-query tile, b, h) ----------
__global__ __launch_bounds__(32) void attn_fwd(
    const unsigned short* __restrict__ q, const unsigned short* __restrict__ k,
    const unsigned short* __restrict__ vt, unsigned short* __restrict__ o)
{
  const int lane = threadIdx.x;
  const int l15  = lane & 15;
  const int hh   = lane >> 4;
  const int qBase = blockIdx.x * 16;
  const int bh = blockIdx.y;
  const int b  = bh >> 4;
  const int h  = bh & 15;

  const unsigned short* Qb = q  + ((size_t)b * T_SEQ) * DMODEL + h * HDIM;
  const unsigned short* Kb = k  + ((size_t)b * T_SEQ) * DMODEL + h * HDIM;
  const unsigned short* Vt = vt + (size_t)bh * HDIM * T_SEQ;   // [64][T]

  // Q A-fragments for head-dim chunks 0..31 and 32..63
  Frag qf[2];
  {
    const unsigned short* Qrow = Qb + (size_t)(qBase + l15) * DMODEL;
#pragma unroll
    for (int c = 0; c < 2; ++c) {
      const int kb = c * 32 + hh * 8;
      qf[c].u[0] = *(const uint4*)(Qrow + kb);
      qf[c].u[1] = *(const uint4*)(Qrow + kb + 16);
    }
  }

  Acc acc[4];
#pragma unroll
  for (int i = 0; i < 4; ++i) acc[i].v = zero8();
  float mrow[8], lrow[8];
#pragma unroll
  for (int r = 0; r < 8; ++r) { mrow[r] = -1e30f; lrow[r] = 0.f; }

  __shared__ unsigned short pT[16][32];       // P tile bounce buffer (C-layout -> A-layout)

  const int kvEnd = qBase + 16;
  for (int j0 = 0; j0 < kvEnd; j0 += 32) {
    // ----- S = (Q @ K^T) * scale, two 16x16 tiles covering 32 keys -----
    Acc s[2];
    s[0].v = zero8(); s[1].v = zero8();
#pragma unroll
    for (int sub = 0; sub < 2; ++sub) {
      int keyRow = j0 + sub * 16 + l15;
      if (keyRow > T_SEQ - 1) keyRow = T_SEQ - 1;           // masked later
      const unsigned short* Krow = Kb + (size_t)keyRow * DMODEL;
#pragma unroll
      for (int c = 0; c < 2; ++c) {
        Frag bf;
        const int kb = c * 32 + hh * 8;
        bf.u[0] = *(const uint4*)(Krow + kb);
        bf.u[1] = *(const uint4*)(Krow + kb + 16);
        s[sub].v = __builtin_amdgcn_wmma_f32_16x16x32_bf16(
            false, qf[c].v, false, bf.v, (short)0, s[sub].v, false, false);
      }
    }
    // scale + causal mask
#pragma unroll
    for (int sub = 0; sub < 2; ++sub)
#pragma unroll
      for (int r = 0; r < 8; ++r) {
        const int key = j0 + sub * 16 + l15;
        const int qu  = qBase + r + hh * 8;
        const float x = s[sub].f[r] * 0.125f;               // 1/sqrt(64)
        s[sub].f[r] = (key <= qu) ? x : -1e30f;
      }
    // ----- online softmax (row stats replicated in each 16-lane half) -----
    float alpha[8];
#pragma unroll
    for (int r = 0; r < 8; ++r) {
      float rm = fmaxf(s[0].f[r], s[1].f[r]);
      rm = fmaxf(rm, __shfl_xor(rm, 1, 32));
      rm = fmaxf(rm, __shfl_xor(rm, 2, 32));
      rm = fmaxf(rm, __shfl_xor(rm, 4, 32));
      rm = fmaxf(rm, __shfl_xor(rm, 8, 32));
      const float nm = fmaxf(mrow[r], rm);
      alpha[r] = __expf(mrow[r] - nm);
      mrow[r] = nm;
      const float p0 = __expf(s[0].f[r] - nm);
      const float p1 = __expf(s[1].f[r] - nm);
      s[0].f[r] = p0; s[1].f[r] = p1;
      float rs = p0 + p1;
      rs += __shfl_xor(rs, 1, 32);
      rs += __shfl_xor(rs, 2, 32);
      rs += __shfl_xor(rs, 4, 32);
      rs += __shfl_xor(rs, 8, 32);
      lrow[r] = lrow[r] * alpha[r] + rs;
    }
#pragma unroll
    for (int i = 0; i < 4; ++i)
#pragma unroll
      for (int r = 0; r < 8; ++r) acc[i].f[r] *= alpha[r];

    // ----- P (C-layout) -> LDS -> A-layout fragment -----
#pragma unroll
    for (int r = 0; r < 8; ++r) {
      const int row = r + hh * 8;
      pT[row][l15]      = f2bf(s[0].f[r]);
      pT[row][16 + l15] = f2bf(s[1].f[r]);
    }
    asm volatile("s_wait_dscnt 0" ::: "memory");  // same-wave LDS RAW safety
    Frag pf;
    pf.u[0] = *(const uint4*)(&pT[l15][hh * 16]);
    pf.u[1] = *(const uint4*)(&pT[l15][hh * 16 + 8]);

    // ----- O += P @ V, V pre-transposed [64][T] so B-fragments are contiguous -----
    int jv = j0 + hh * 16;
    if (jv > T_SEQ - 16) jv = T_SEQ - 16;         // masked (P==0) region, clamp load
#pragma unroll
    for (int nch = 0; nch < 4; ++nch) {
      const unsigned short* Vrow = Vt + (size_t)(nch * 16 + l15) * T_SEQ + jv;
      Frag vf;
      vf.u[0] = *(const uint4*)(Vrow);
      vf.u[1] = *(const uint4*)(Vrow + 8);
      acc[nch].v = __builtin_amdgcn_wmma_f32_16x16x32_bf16(
          false, pf.v, false, vf.v, (short)0, acc[nch].v, false, false);
    }
  }

  // normalize and write o (bf16 row-major [N, D])
#pragma unroll
  for (int nch = 0; nch < 4; ++nch)
#pragma unroll
    for (int r = 0; r < 8; ++r) {
      const int row = qBase + r + hh * 8;
      const int col = h * HDIM + nch * 16 + l15;
      const float val = acc[nch].f[r] / lrow[r];
      o[((size_t)(b * T_SEQ + row)) * DMODEL + col] = f2bf(val);
    }
}

// ---------- LayerNorm: f32 in -> bf16 out, one block per row ----------
__global__ __launch_bounds__(256) void ln_bf16(
    const float* __restrict__ x, const float* __restrict__ g,
    const float* __restrict__ bta, unsigned short* __restrict__ out)
{
  __shared__ float a1[256];
  __shared__ float a2[256];
  const int row = blockIdx.x;
  const int tid = threadIdx.x;
  const float4 v = ((const float4*)(x + (size_t)row * DMODEL))[tid];
  a1[tid] = v.x + v.y + v.z + v.w;
  a2[tid] = v.x * v.x + v.y * v.y + v.z * v.z + v.w * v.w;
  __syncthreads();
  for (int s = 128; s > 0; s >>= 1) {
    if (tid < s) { a1[tid] += a1[tid + s]; a2[tid] += a2[tid + s]; }
    __syncthreads();
  }
  const float mean = a1[0] * (1.0f / DMODEL);
  const float var  = a2[0] * (1.0f / DMODEL) - mean * mean;
  const float rstd = rsqrtf(var + 1e-5f);
  const float4 gg = ((const float4*)g)[tid];
  const float4 bb = ((const float4*)bta)[tid];
  ushort4 r;
  r.x = f2bf((v.x - mean) * rstd * gg.x + bb.x);
  r.y = f2bf((v.y - mean) * rstd * gg.y + bb.y);
  r.z = f2bf((v.z - mean) * rstd * gg.z + bb.z);
  r.w = f2bf((v.w - mean) * rstd * gg.w + bb.w);
  ((ushort4*)(out + (size_t)row * DMODEL))[tid] = r;
}

// ---------- embedding: x = tok_emb[ctx] + pos_emb ----------
__global__ __launch_bounds__(256) void embed_k(
    const int* __restrict__ ctx, const float* __restrict__ tok,
    const float* __restrict__ pos, float* __restrict__ x)
{
  const size_t i = (size_t)blockIdx.x * 256 + threadIdx.x;
  const int row = (int)(i >> 10);
  const int col = (int)(i & (DMODEL - 1));
  const int t   = row & (T_SEQ - 1);
  x[i] = tok[(size_t)ctx[row] * DMODEL + col] + pos[(size_t)t * DMODEL + col];
}

// ---------- weight convert+transpose: f32 [K,N] -> bf16 [N,K] (tiled via LDS) ----------
__global__ __launch_bounds__(256) void wconv(
    const float* __restrict__ W, unsigned short* __restrict__ Wt, int K, int N)
{
  __shared__ unsigned short tile[32][33];
  const int n0 = blockIdx.x * 32;
  const int k0 = blockIdx.y * 32;
  const int tx = threadIdx.x;   // 0..31
  const int ty = threadIdx.y;   // 0..7
  for (int i = ty; i < 32; i += 8)
    tile[i][tx] = f2bf(W[(size_t)(k0 + i) * N + n0 + tx]);
  __syncthreads();
  for (int i = ty; i < 32; i += 8)
    Wt[(size_t)(n0 + i) * K + k0 + tx] = tile[tx][i];
}

// ---------- orchestration ----------
extern "C" void kernel_launch(void* const* d_in, const int* in_sizes, int n_in,
                              void* d_out, int out_size, void* d_ws, size_t ws_size,
                              hipStream_t stream)
{
  (void)in_sizes; (void)n_in; (void)out_size; (void)ws_size;
  const int*   ctx  = (const int*)  d_in[0];
  const float* tok  = (const float*)d_in[1];
  const float* pos  = (const float*)d_in[2];
  const float* Wq   = (const float*)d_in[3];
  const float* Wk   = (const float*)d_in[4];
  const float* Wv   = (const float*)d_in[5];
  const float* Wo   = (const float*)d_in[6];
  const float* bo   = (const float*)d_in[7];
  const float* ln1s = (const float*)d_in[8];
  const float* ln1b = (const float*)d_in[9];
  const float* W1   = (const float*)d_in[10];
  const float* b1   = (const float*)d_in[11];
  const float* W2   = (const float*)d_in[12];
  const float* b2   = (const float*)d_in[13];
  const float* ln2s = (const float*)d_in[14];
  const float* ln2b = (const float*)d_in[15];
  const float* lnfs = (const float*)d_in[16];
  const float* lnfb = (const float*)d_in[17];
  const float* Wout = (const float*)d_in[18];
  const float* bout = (const float*)d_in[19];
  float* logits = (float*)d_out;

  // workspace layout
  char* w = (char*)d_ws;
  float*          xf  = (float*)w;          w += (size_t)NROWS * DMODEL * 4;       // x (f32)
  unsigned short* hb  = (unsigned short*)w; w += (size_t)NROWS * DMODEL * 2;       // ln out (bf16)
  unsigned short* qb  = (unsigned short*)w; w += (size_t)NROWS * DMODEL * 2;
  unsigned short* kbuf= (unsigned short*)w; w += (size_t)NROWS * DMODEL * 2;
  unsigned short* vtb = (unsigned short*)w; w += (size_t)NROWS * DMODEL * 2;       // V transposed
  unsigned short* ob  = (unsigned short*)w; w += (size_t)NROWS * DMODEL * 2;
  unsigned short* f1b = (unsigned short*)w; w += (size_t)NROWS * 4 * DMODEL * 2;   // relu out
  unsigned short* wts = (unsigned short*)w;                                        // bf16 weights

  const size_t M1 = 1024ull * 1024ull;   // 1M elements
  const dim3 cb(32, 8);

  // --- convert all weights to bf16, transposed [N][K] ---
  for (int l = 0; l < 4; ++l) {
    unsigned short* base = wts + (size_t)l * 12 * M1;
    wconv<<<dim3(32, 32),  cb, 0, stream>>>(Wq + (size_t)l * M1,     base + 0 * M1, 1024, 1024);
    wconv<<<dim3(32, 32),  cb, 0, stream>>>(Wk + (size_t)l * M1,     base + 1 * M1, 1024, 1024);
    wconv<<<dim3(32, 32),  cb, 0, stream>>>(Wv + (size_t)l * M1,     base + 2 * M1, 1024, 1024);
    wconv<<<dim3(32, 32),  cb, 0, stream>>>(Wo + (size_t)l * M1,     base + 3 * M1, 1024, 1024);
    wconv<<<dim3(128, 32), cb, 0, stream>>>(W1 + (size_t)l * 4 * M1, base + 4 * M1, 1024, 4096);
    wconv<<<dim3(32, 128), cb, 0, stream>>>(W2 + (size_t)l * 4 * M1, base + 8 * M1, 4096, 1024);
  }
  unsigned short* woutT = wts + 48 * M1;
  wconv<<<dim3(1000, 32), cb, 0, stream>>>(Wout, woutT, 1024, 32000);

  // --- embeddings ---
  embed_k<<<(NROWS * DMODEL) / 256, 256, 0, stream>>>(ctx, tok, pos, xf);

  // --- transformer layers ---
  for (int l = 0; l < 4; ++l) {
    unsigned short* base = wts + (size_t)l * 12 * M1;
    ln_bf16<<<NROWS, 256, 0, stream>>>(xf, ln1s + l * 1024, ln1b + l * 1024, hb);
    gemm_bf16<<<dim3(16, 32), 256, 0, stream>>>(hb, base + 0 * M1, nullptr, nullptr, qb,
                                                NROWS, 1024, 1024, GF_OUTB);
    gemm_bf16<<<dim3(16, 32), 256, 0, stream>>>(hb, base + 1 * M1, nullptr, nullptr, kbuf,
                                                NROWS, 1024, 1024, GF_OUTB);
    gemm_bf16<<<dim3(16, 32), 256, 0, stream>>>(hb, base + 2 * M1, nullptr, nullptr, vtb,
                                                NROWS, 1024, 1024, GF_VT);
    attn_fwd<<<dim3(T_SEQ / 16, 32), 32, 0, stream>>>(qb, kbuf, vtb, ob);
    gemm_bf16<<<dim3(16, 32), 256, 0, stream>>>(ob, base + 3 * M1, bo + l * 1024, xf, nullptr,
                                                NROWS, 1024, 1024, GF_BIAS | GF_RES | GF_OUTF);
    ln_bf16<<<NROWS, 256, 0, stream>>>(xf, ln2s + l * 1024, ln2b + l * 1024, hb);
    gemm_bf16<<<dim3(64, 32), 256, 0, stream>>>(hb, base + 4 * M1, b1 + l * 4096, nullptr, f1b,
                                                NROWS, 4096, 1024, GF_BIAS | GF_RELU | GF_OUTB);
    gemm_bf16<<<dim3(16, 32), 256, 0, stream>>>(f1b, base + 8 * M1, b2 + l * 1024, xf, nullptr,
                                                NROWS, 1024, 4096, GF_BIAS | GF_RES | GF_OUTF);
  }

  // --- final LN + LM head ---
  ln_bf16<<<NROWS, 256, 0, stream>>>(xf, lnfs, lnfb, hb);
  gemm_bf16<<<dim3(500, 32), 256, 0, stream>>>(hb, woutT, bout, logits, nullptr,
                                               NROWS, 32000, 1024, GF_BIAS | GF_OUTF);
}
